// SBDHT_69965017252054
// MI455X (gfx1250) — matrix operand order repaired and checked by reading, not compiled, not run on hardware
//
#include <hip/hip_runtime.h>
#include <hip/hip_bf16.h>

typedef __bf16 bf16;
typedef __attribute__((ext_vector_type(16))) __bf16 v16bf;
typedef __attribute__((ext_vector_type(8)))  __bf16 v8bf;
typedef __attribute__((ext_vector_type(8)))  float  v8f;

#define HIDDEN   2048
#define NB       64        // blocks per row
#define BLK      32
#define ROWS     16        // rows per workgroup tile (WMMA M)
#define NTHREADS 512       // 16 waves of 32
#define XROW     2064      // padded halves per x/inp row (2048+16)
#define TROW     72        // padded halves per (j,b) m-row (64+8)
#define TJ       (16*TROW) // halves per j-plane
#define HROW     72        // padded H64 row

__device__ __forceinline__ v16bf cat8(v8bf lo, v8bf hi) {
    return __builtin_shufflevector(lo, hi, 0,1,2,3,4,5,6,7,8,9,10,11,12,13,14,15);
}

// ---------------------------------------------------------------------------
// Prep: fold H32 into each 32x32 block weight, emit bf16 row-major V[j][k].
// V[i][s][n][j][k] = sum_l (-1)^popc(j&l) * W[i][s][n][l][k]
// grid = 4*2*64 blocks, 1024 threads (one output element each).
// ---------------------------------------------------------------------------
__global__ __launch_bounds__(1024) void SBDHT_prep_kernel(
    const float* __restrict__ w, bf16* __restrict__ vout)
{
    const int blk = blockIdx.x;            // ((it*2)+side)*64 + n
    const int j = threadIdx.x >> 5;
    const int k = threadIdx.x & 31;
    const float* wb = w + (long)blk * 1024;
    float acc = 0.f;
    #pragma unroll
    for (int l = 0; l < 32; ++l) {
        float s = (__builtin_popcount(j & l) & 1) ? -1.f : 1.f;
        acc += s * wb[l * 32 + k];
    }
    vout[(long)blk * 1024 + j * 32 + k] = (bf16)acc;
}

// ---------------------------------------------------------------------------
// Main: one workgroup = 16 rows x 2048, state resident in LDS for 4 iters.
// Per half-iteration:
//   step1 (per block n):  t[b,n,j] = sum_k x[b,n,k] * V_n[j,k]
//        GEMM M=j(32), N=b(16), K=32  -> D written to tbuf[j][b][m=n]
//   step2 (per j):        z[b,n,j] = sum_m t[b,m,j] * H64[m,n]
//        GEMM M=b(16), N=n(64), K=m(64) -> blended, written back to xbuf
// ---------------------------------------------------------------------------
__global__ __launch_bounds__(NTHREADS) void SBDHT_main_kernel(
    const float* __restrict__ x_in, const bf16* __restrict__ Vg,
    const float* __restrict__ bias, float* __restrict__ out)
{
    // LDS carve-up (halves): xbuf 16*2064, ibuf 16*2064, tbuf 32*16*72, hbuf 64*72
    __shared__ bf16 smem[2 * ROWS * XROW + 32 * TJ + 64 * HROW]; // 215,040 bytes
    bf16* xbuf = smem;                       // [16][2064]
    bf16* ibuf = smem + ROWS * XROW;         // [16][2064]
    bf16* tbuf = ibuf + ROWS * XROW;         // [32][16][72]
    bf16* hbuf = tbuf + 32 * TJ;             // [64][72]

    const int tid  = threadIdx.x;
    const int lane = tid & 31;
    const int wave = tid >> 5;
    const int g    = lane >> 4;              // lane half-group
    const int lb   = lane & 15;
    const long row0 = (long)blockIdx.x * ROWS;

    // Build H64 (+/-1, exact in bf16)
    for (int idx = tid; idx < 64 * 64; idx += NTHREADS) {
        int n = idx >> 6, m = idx & 63;
        hbuf[n * HROW + m] = (__builtin_popcount(n & m) & 1) ? (bf16)(-1.0f)
                                                             : (bf16)( 1.0f);
    }
    // Load input tile once (coalesced), keep bf16 copies of state + inp
    const float4* gin = (const float4*)(x_in + row0 * HIDDEN);
    for (int i = tid; i < ROWS * HIDDEN / 4; i += NTHREADS) {
        int b  = i >> 9;               // 512 float4 per row
        int h4 = i & 511;
        float4 v = gin[b * 512 + h4];
        int o = b * XROW + h4 * 4;
        xbuf[o+0] = (bf16)v.x; xbuf[o+1] = (bf16)v.y;
        xbuf[o+2] = (bf16)v.z; xbuf[o+3] = (bf16)v.w;
        ibuf[o+0] = (bf16)v.x; ibuf[o+1] = (bf16)v.y;
        ibuf[o+2] = (bf16)v.z; ibuf[o+3] = (bf16)v.w;
    }

    for (int phase = 0; phase < 8; ++phase) {       // 4 iters x 2 sides
        const int side = phase & 1;
        __syncthreads();

        // ---- step 1: each wave owns 4 blocks ----
        const bf16* Vbase = Vg + (long)phase * (64 * 1024);
        #pragma unroll
        for (int bi = 0; bi < 4; ++bi) {
            const int n = wave * 4 + bi;
            // B operand: x rows, lane -> N=b, K runs {g*8, 16+g*8} from xbuf
            const bf16* xp = xbuf + lb * XROW + n * 32 + g * 8;
            v16bf Bm = cat8(*(const v8bf*)xp, *(const v8bf*)(xp + 16));
            #pragma unroll
            for (int jc = 0; jc < 2; ++jc) {
                // A operand: V rows (lane -> M=j), K runs {g*8, 16+g*8}
                const bf16* vr = Vbase + n * 1024 + (jc * 16 + lb) * 32 + g * 8;
                v16bf Am = cat8(*(const v8bf*)vr, *(const v8bf*)(vr + 16));
                v8f acc = {};
                acc = __builtin_amdgcn_wmma_f32_16x16x32_bf16(
                        false, Am, false, Bm, (short)0, acc, false, false);
                // D: VGPR r -> j' = r + 8g, lane -> b; scatter into tbuf[j][b][n]
                #pragma unroll
                for (int r = 0; r < 8; ++r) {
                    int j = jc * 16 + r + g * 8;
                    tbuf[j * TJ + lb * TROW + n] = (bf16)acc[r];
                }
            }
        }
        __syncthreads();

        // ---- step 2: each wave owns 2 j-planes ----
        #pragma unroll
        for (int ji = 0; ji < 2; ++ji) {
            const int j = wave + ji * 16;
            const bf16* tb = tbuf + j * TJ + lb * TROW;
            #pragma unroll
            for (int nc = 0; nc < 4; ++nc) {
                v8f acc = {};
                #pragma unroll
                for (int q = 0; q < 2; ++q) {
                    // A: t rows (lane -> M=b), K=m runs {q*32+g*8, q*32+16+g*8}
                    const bf16* ta = tb + q * 32 + g * 8;
                    v16bf Am = cat8(*(const v8bf*)ta, *(const v8bf*)(ta + 16));
                    // B: H64 (lane -> N=n), m runs {q*32+g*8, q*32+16+g*8}
                    const bf16* hp = hbuf + (nc * 16 + lb) * HROW + q * 32 + g * 8;
                    v16bf Bm = cat8(*(const v8bf*)hp, *(const v8bf*)(hp + 16));
                    acc = __builtin_amdgcn_wmma_f32_16x16x32_bf16(
                            false, Am, false, Bm, (short)0, acc, false, false);
                }
                const int n = nc * 16 + lb;
                const int h = n * 32 + j;
                #pragma unroll
                for (int r = 0; r < 8; ++r) {
                    int b = r + g * 8;
                    float z = acc[r];
                    if (side) z = 0.9f * z + 0.1f * (float)ibuf[b * XROW + h];
                    xbuf[b * XROW + h] = (bf16)z;
                }
            }
        }
    }
    __syncthreads();

    // Output: coalesced f32 store + bias
    for (int i = tid; i < ROWS * HIDDEN; i += NTHREADS) {
        int b = i >> 11, h = i & 2047;
        out[(row0 + b) * HIDDEN + h] = (float)xbuf[b * XROW + h] + bias[h];
    }
}

extern "C" void kernel_launch(void* const* d_in, const int* in_sizes, int n_in,
                              void* d_out, int out_size, void* d_ws, size_t ws_size,
                              hipStream_t stream) {
    const float* hs   = (const float*)d_in[0];   // (4,2048,2048)
    const float* w    = (const float*)d_in[1];   // (4,2,64,32,32)
    const float* bias = (const float*)d_in[2];   // (2048,)
    float* out = (float*)d_out;
    bf16* Vg = (bf16*)d_ws;                      // 4*2*64*32*32 bf16 = 1 MB

    SBDHT_prep_kernel<<<4 * 2 * 64, 1024, 0, stream>>>(w, Vg);
    SBDHT_main_kernel<<<(4 * 2048) / ROWS, NTHREADS, 0, stream>>>(hs, Vg, bias, out);
}
